// MultiHeadAttention_31473520345937
// MI455X (gfx1250) — matrix operand order repaired and checked by reading, not compiled, not run on hardware
//
#include <hip/hip_runtime.h>
#include <hip/hip_fp16.h>

#define DEV __device__ __forceinline__

typedef __attribute__((ext_vector_type(16))) _Float16 v16h;
typedef __attribute__((ext_vector_type(8)))  _Float16 v8h;
typedef __attribute__((ext_vector_type(4)))  _Float16 v4h;
typedef __attribute__((ext_vector_type(8)))  float    v8f;

constexpr int Bc  = 2;
constexpr int Lc  = 2048;
constexpr int Dc  = 1024;
constexpr int Hc  = 16;
constexpr int DHc = 64;
constexpr int EROWS = 2048 + 96;   // zero-padded tail for skew window over-read

// ---------------------------------------------------------------- helpers ---
DEV v8f vzero() {
  v8f z;
#pragma unroll
  for (int i = 0; i < 8; ++i) z[i] = 0.f;
  return z;
}

DEV v16h make_frag(v8h lo, v8h hi) {
  v16h r;
#pragma unroll
  for (int i = 0; i < 8; ++i) { r[i] = lo[i]; r[i + 8] = hi[i]; }
  return r;
}

// A-matrix fragment (16x32 f16): lane l -> row M=l&15, group g=l>>4.
// halves [0..7] = K = g*8..g*8+7 ; halves [8..15] = K = 16+g*8..
DEV v16h load_a_frag(const _Float16* base, int ld) {
  const int lane = threadIdx.x & 31;
  const int m = lane & 15, g = lane >> 4;
  const _Float16* p = base + (size_t)m * ld + g * 8;
  v8h lo = *(const v8h*)p;
  v8h hi = *(const v8h*)(p + 16);
  return make_frag(lo, hi);
}

// B-matrix fragment (32x16 f16) stored as [N][K] with row-stride ld:
// lane l -> col N=l&15, g=l>>4; halves [0..7] = K = g*8.. ; [8..15] = K = 16+g*8..
DEV v16h load_b_frag(const _Float16* base, int ld) {
  const int lane = threadIdx.x & 31;
  const int n = lane & 15, g = lane >> 4;
  const _Float16* p = base + (size_t)n * ld + g * 8;
  v8h lo = *(const v8h*)p;
  v8h hi = *(const v8h*)(p + 16);
  return make_frag(lo, hi);
}

DEV v8f wmma16(v16h a, v16h b, v8f c) {
  return __builtin_amdgcn_wmma_f32_16x16x32_f16(false, a, false, b, (short)0, c,
                                                false, false);
}

// ------------------------------------------------------------ kernel 1: e ---
__global__ void cvt_e_kernel(const float* __restrict__ rel,
                             _Float16* __restrict__ e) {
  int idx = blockIdx.x * blockDim.x + threadIdx.x;
  if (idx >= EROWS * DHc) return;
  int m = idx / DHc, d = idx - m * DHc;
  float v = (m < Lc) ? rel[(size_t)(m + 1) * DHc + d] : 0.f;
  e[idx] = (_Float16)v;
}

// --------------------------------------------------- kernel 2: QKV project ---
// C(4096x1024) = A(f32) @ W(f32) + bias ; out f16 head-split (V transposed).
__global__ __launch_bounds__(256) void qkv_proj_kernel(
    const float* __restrict__ qin, const float* __restrict__ kin,
    const float* __restrict__ vin,
    const float* __restrict__ Wq, const float* __restrict__ bq,
    const float* __restrict__ Wk, const float* __restrict__ bk,
    const float* __restrict__ Wv, const float* __restrict__ bv,
    _Float16* __restrict__ qf, _Float16* __restrict__ kf,
    _Float16* __restrict__ vt) {
  constexpr int LDA = 40;                       // halves (pad vs bank conflict)
  __shared__ _Float16 As[128 * LDA];
  __shared__ _Float16 Ws[128 * LDA];            // [n][k] (pre-transposed)

  const int z = blockIdx.z;
  const float* A    = z == 0 ? qin : (z == 1 ? kin : vin);
  const float* W    = z == 0 ? Wq  : (z == 1 ? Wk  : Wv);
  const float* bias = z == 0 ? bq  : (z == 1 ? bk  : bv);
  const int r0 = blockIdx.x * 128;
  const int c0 = blockIdx.y * 128;
  const int tid = threadIdx.x;
  const int wid = tid >> 5, lane = tid & 31;
  const int wm = wid >> 1, wn = wid & 1;        // wave tile: rows wm*32, cols wn*64
  const int g = lane >> 4, nn = lane & 15;

  v8f acc[2][4];
#pragma unroll
  for (int i = 0; i < 2; ++i)
#pragma unroll
    for (int j = 0; j < 4; ++j) acc[i][j] = vzero();

  for (int kt = 0; kt < Dc / 32; ++kt) {
    // stage A tile 128x32 (f32 -> f16)
#pragma unroll
    for (int it = 0; it < 4; ++it) {
      int idx4 = it * 256 + tid;                 // 1024 float4
      int m = idx4 >> 3, kq = idx4 & 7;
      float4 f = *(const float4*)(A + (size_t)(r0 + m) * Dc + kt * 32 + kq * 4);
      v4h h;
      h[0] = (_Float16)f.x; h[1] = (_Float16)f.y;
      h[2] = (_Float16)f.z; h[3] = (_Float16)f.w;
      *(v4h*)(As + m * LDA + kq * 4) = h;
    }
    // stage W tile transposed: Ws[n][k] = W[kt*32+k][c0+n]
#pragma unroll
    for (int it = 0; it < 4; ++it) {
      int idx4 = it * 256 + tid;
      int kk = idx4 >> 5, n4 = (idx4 & 31) * 4;
      float4 f = *(const float4*)(W + (size_t)(kt * 32 + kk) * Dc + c0 + n4);
      Ws[(n4 + 0) * LDA + kk] = (_Float16)f.x;
      Ws[(n4 + 1) * LDA + kk] = (_Float16)f.y;
      Ws[(n4 + 2) * LDA + kk] = (_Float16)f.z;
      Ws[(n4 + 3) * LDA + kk] = (_Float16)f.w;
    }
    __syncthreads();

    v16h af[2], bf[4];
#pragma unroll
    for (int rt = 0; rt < 2; ++rt)
      af[rt] = load_a_frag(As + (wm * 32 + rt * 16) * LDA, LDA);
#pragma unroll
    for (int ct = 0; ct < 4; ++ct)
      bf[ct] = load_b_frag(Ws + (wn * 64 + ct * 16) * LDA, LDA);
#pragma unroll
    for (int rt = 0; rt < 2; ++rt)
#pragma unroll
      for (int ct = 0; ct < 4; ++ct)
        acc[rt][ct] = wmma16(af[rt], bf[ct], acc[rt][ct]);
    __syncthreads();
  }

  // epilogue: bias, cvt f16, head-split scatter. z-branch hoisted; V output is
  // transposed [bh][dh][l], so a lane's 8 rows are 8 consecutive l -> one v8h.
  if (z == 2) {
#pragma unroll
    for (int rt = 0; rt < 2; ++rt)
#pragma unroll
      for (int ct = 0; ct < 4; ++ct) {
        int col = c0 + wn * 64 + ct * 16 + nn;
        int h = col >> 6, dh = col & 63;
        float bsc = bias[col];
        int row0 = r0 + wm * 32 + rt * 16 + 8 * g;   // 8-aligned, same batch
        int bb = row0 >> 11, l0 = row0 & 2047;
        v8h pk;
#pragma unroll
        for (int r = 0; r < 8; ++r) pk[r] = (_Float16)(acc[rt][ct][r] + bsc);
        *(v8h*)(vt + ((size_t)((bb * Hc + h) * DHc + dh)) * Lc + l0) = pk;
      }
  } else {
    _Float16* dst = (z == 0) ? qf : kf;
#pragma unroll
    for (int rt = 0; rt < 2; ++rt)
#pragma unroll
      for (int ct = 0; ct < 4; ++ct) {
        int col = c0 + wn * 64 + ct * 16 + nn;
        int h = col >> 6, dh = col & 63;
        float bsc = bias[col];
        int row0 = r0 + wm * 32 + rt * 16 + 8 * g;
        int bb = row0 >> 11, l0 = row0 & 2047;
        _Float16* p = dst + ((size_t)(bb * Hc + h) * Lc + l0) * DHc + dh;
#pragma unroll
        for (int r = 0; r < 8; ++r)
          p[(size_t)r * DHc] = (_Float16)(acc[rt][ct][r] + bsc);
      }
  }
}

// --------------------------------------------- kernel 3: fused flash attn ---
__global__ __launch_bounds__(256) void flash_attn_kernel(
    const _Float16* __restrict__ qf, const _Float16* __restrict__ kf,
    const _Float16* __restrict__ vt, const _Float16* __restrict__ e,
    _Float16* __restrict__ ao) {
  constexpr int RLD = 100;                       // floats per R row
  constexpr int PLD = 72;                        // halves per P row
  __shared__ float lds[8 * 16 * RLD];            // 51200 B, per-wave strips

  const int r0 = blockIdx.x * 128;
  const int bh = blockIdx.y;
  const int bb = bh >> 4, h = bh & 15;
  const int wid = threadIdx.x >> 5, lane = threadIdx.x & 31;
  const int rw = r0 + wid * 16;                  // this wave's 16 query rows
  const int g = lane >> 4, nn = lane & 15;
  float*     Rbuf = lds + wid * 16 * RLD;
  _Float16*  Pbuf = (_Float16*)Rbuf;             // reused (same-wave, in-order DS)

  const _Float16* qbase = qf + ((size_t)bh * Lc + rw) * DHc;
  const _Float16* kbase = kf + (size_t)bh * Lc * DHc;
  const _Float16* vbase = vt + (size_t)bh * DHc * Lc;

  v16h aq0 = load_a_frag(qbase, DHc);            // dh 0..31
  v16h aq1 = load_a_frag(qbase + 32, DHc);       // dh 32..63

  v8f o[4];
#pragma unroll
  for (int i = 0; i < 4; ++i) o[i] = vzero();
  float mrow[8], lrow[8];
#pragma unroll
  for (int r = 0; r < 8; ++r) { mrow[r] = -1e30f; lrow[r] = 0.f; }

  for (int jb = 0; jb < Lc / 64; ++jb) {
    const int j0 = jb * 64;

    // prefetch next chunk's K (8 KB) and V columns into cache while this
    // chunk's WMMAs run (global_prefetch_b8)
    if (jb + 1 < Lc / 64) {
      const _Float16* knext = kbase + (size_t)(j0 + 64) * DHc;
      __builtin_prefetch(knext + (size_t)lane * 128, 0, 3);
      const _Float16* vnext = vbase + j0 + 64;
      __builtin_prefetch(vnext + (size_t)(lane * 2) * Lc, 0, 3);
      __builtin_prefetch(vnext + (size_t)(lane * 2 + 1) * Lc, 0, 3);
    }

    // ---- S = Q * K^T (raw) ----
    v8f s[4];
#pragma unroll
    for (int ct = 0; ct < 4; ++ct) {
      const _Float16* kb = kbase + (size_t)(j0 + ct * 16) * DHc;
      v16h b0 = load_b_frag(kb, DHc);
      v16h b1 = load_b_frag(kb + 32, DHc);
      s[ct] = wmma16(aq1, b1, wmma16(aq0, b0, vzero()));
    }

    // ---- relative bias (wave-uniform branch -> EXEC stays all ones) ----
    if (j0 <= rw + 15) {
      const int m_lo = 2047 - (rw + 15) + j0;
      const int m0 = m_lo & ~15;
      v8f rr[6];
#pragma unroll
      for (int rt = 0; rt < 6; ++rt) {
        const _Float16* eb = e + (size_t)(m0 + rt * 16) * DHc;
        v16h b0 = load_b_frag(eb, DHc);
        v16h b1 = load_b_frag(eb + 32, DHc);
        rr[rt] = wmma16(aq1, b1, wmma16(aq0, b0, vzero()));
      }
      // stage R (C-layout) to per-wave LDS strip
#pragma unroll
      for (int rt = 0; rt < 6; ++rt)
#pragma unroll
        for (int r = 0; r < 8; ++r)
          Rbuf[(r + 8 * g) * RLD + rt * 16 + nn] = rr[rt][r];
      __builtin_amdgcn_sched_barrier(0);
      asm volatile("" ::: "memory");
      // Toeplitz-skewed gather: srel[i][j] = R[i][2047-i+j - m0] for j<=i
#pragma unroll
      for (int ct = 0; ct < 4; ++ct) {
        int j = j0 + ct * 16 + nn;
#pragma unroll
        for (int r = 0; r < 8; ++r) {
          int iloc = r + 8 * g;
          int i = rw + iloc;
          if (j <= i) s[ct][r] += Rbuf[iloc * RLD + (2047 - i + j - m0)];
        }
      }
      __builtin_amdgcn_sched_barrier(0);
      asm volatile("" ::: "memory");
    }

    // ---- scale + online softmax ----
#pragma unroll
    for (int ct = 0; ct < 4; ++ct) s[ct] = s[ct] * 0.125f;
#pragma unroll
    for (int r = 0; r < 8; ++r) {
      float mx = fmaxf(fmaxf(s[0][r], s[1][r]), fmaxf(s[2][r], s[3][r]));
#pragma unroll
      for (int off = 1; off < 16; off <<= 1)
        mx = fmaxf(mx, __shfl_xor(mx, off, 32));
      float mnew = fmaxf(mrow[r], mx);
      float alpha = __expf(mrow[r] - mnew);
      mrow[r] = mnew;
      float ls = 0.f;
#pragma unroll
      for (int ct = 0; ct < 4; ++ct) {
        float p = __expf(s[ct][r] - mnew);
        s[ct][r] = p;
        ls += p;
      }
#pragma unroll
      for (int off = 1; off < 16; off <<= 1) ls += __shfl_xor(ls, off, 32);
      lrow[r] = lrow[r] * alpha + ls;
#pragma unroll
      for (int dt = 0; dt < 4; ++dt) o[dt][r] *= alpha;
    }

    // ---- P: C-layout -> A-layout via LDS, then O += P * V ----
#pragma unroll
    for (int ct = 0; ct < 4; ++ct)
#pragma unroll
      for (int r = 0; r < 8; ++r)
        Pbuf[(r + 8 * g) * PLD + ct * 16 + nn] = (_Float16)s[ct][r];
    __builtin_amdgcn_sched_barrier(0);
    asm volatile("" ::: "memory");
    v16h ap0 = load_a_frag(Pbuf, PLD);           // j local 0..31
    v16h ap1 = load_a_frag(Pbuf + 32, PLD);      // j local 32..63
    __builtin_amdgcn_sched_barrier(0);
    asm volatile("" ::: "memory");
#pragma unroll
    for (int dt = 0; dt < 4; ++dt) {
      const _Float16* vb = vbase + (size_t)(dt * 16) * Lc + j0;
      v16h b0 = load_b_frag(vb, Lc);
      v16h b1 = load_b_frag(vb + 32, Lc);
      o[dt] = wmma16(ap1, b1, wmma16(ap0, b0, o[dt]));
    }
  }

  // ---- normalize + write attn output (B,L,D) f16 ----
  {
    int i0 = rw + 8 * g;
    _Float16* p = ao + ((size_t)bb * Lc + i0) * Dc + h * DHc + nn;
#pragma unroll
    for (int dt = 0; dt < 4; ++dt)
#pragma unroll
      for (int r = 0; r < 8; ++r) {
        float val = o[dt][r] / lrow[r];
        p[(size_t)r * Dc + dt * 16] = (_Float16)val;
      }
  }
}

// ------------------------------------------------- kernel 4: out project ----
__global__ __launch_bounds__(256) void out_proj_kernel(
    const _Float16* __restrict__ A, const float* __restrict__ W,
    const float* __restrict__ bias, float* __restrict__ out) {
  constexpr int LDA = 40;
  __shared__ _Float16 As[128 * LDA];
  __shared__ _Float16 Ws[128 * LDA];
  const int r0 = blockIdx.x * 128;
  const int c0 = blockIdx.y * 128;
  const int tid = threadIdx.x;
  const int wid = tid >> 5, lane = tid & 31;
  const int wm = wid >> 1, wn = wid & 1;
  const int g = lane >> 4, nn = lane & 15;

  v8f acc[2][4];
#pragma unroll
  for (int i = 0; i < 2; ++i)
#pragma unroll
    for (int j = 0; j < 4; ++j) acc[i][j] = vzero();

  for (int kt = 0; kt < Dc / 32; ++kt) {
#pragma unroll
    for (int it = 0; it < 2; ++it) {             // 512 v8h cover 128x32 halves
      int idx8 = it * 256 + tid;
      int m = idx8 >> 2, k8 = idx8 & 3;
      v8h hv = *(const v8h*)(A + (size_t)(r0 + m) * Dc + kt * 32 + k8 * 8);
      *(v8h*)(As + m * LDA + k8 * 8) = hv;
    }
#pragma unroll
    for (int it = 0; it < 4; ++it) {
      int idx4 = it * 256 + tid;
      int kk = idx4 >> 5, n4 = (idx4 & 31) * 4;
      float4 f = *(const float4*)(W + (size_t)(kt * 32 + kk) * Dc + c0 + n4);
      Ws[(n4 + 0) * LDA + kk] = (_Float16)f.x;
      Ws[(n4 + 1) * LDA + kk] = (_Float16)f.y;
      Ws[(n4 + 2) * LDA + kk] = (_Float16)f.z;
      Ws[(n4 + 3) * LDA + kk] = (_Float16)f.w;
    }
    __syncthreads();
    v16h af[2], bf[4];
#pragma unroll
    for (int rt = 0; rt < 2; ++rt)
      af[rt] = load_a_frag(As + (wm * 32 + rt * 16) * LDA, LDA);
#pragma unroll
    for (int ct = 0; ct < 4; ++ct)
      bf[ct] = load_b_frag(Ws + (wn * 64 + ct * 16) * LDA, LDA);
#pragma unroll
    for (int rt = 0; rt < 2; ++rt)
#pragma unroll
      for (int ct = 0; ct < 4; ++ct)
        acc[rt][ct] = wmma16(af[rt], bf[ct], acc[rt][ct]);
    __syncthreads();
  }

#pragma unroll
  for (int rt = 0; rt < 2; ++rt)
#pragma unroll
    for (int ct = 0; ct < 4; ++ct) {
      int col = c0 + wn * 64 + ct * 16 + nn;
      float bsc = bias[col];
      int row0 = r0 + wm * 32 + rt * 16 + 8 * g;
      float* p = out + (size_t)row0 * Dc + col;
#pragma unroll
      for (int r = 0; r < 8; ++r) p[(size_t)r * Dc] = acc[rt][ct][r] + bsc;
    }
}

// ------------------------------------------------------------------ launch --
extern "C" void kernel_launch(void* const* d_in, const int* in_sizes, int n_in,
                              void* d_out, int out_size, void* d_ws,
                              size_t ws_size, hipStream_t stream) {
  (void)in_sizes; (void)n_in; (void)out_size; (void)ws_size;
  const float* query = (const float*)d_in[0];
  const float* key   = (const float*)d_in[1];
  const float* value = (const float*)d_in[2];
  const float* Wq = (const float*)d_in[3];  const float* bq = (const float*)d_in[4];
  const float* Wk = (const float*)d_in[5];  const float* bk = (const float*)d_in[6];
  const float* Wv = (const float*)d_in[7];  const float* bv = (const float*)d_in[8];
  const float* Wo = (const float*)d_in[9];  const float* bo = (const float*)d_in[10];
  const float* rel = (const float*)d_in[11];

  char* ws = (char*)d_ws;
  _Float16* qf = (_Float16*)(ws);                    //  8 MB
  _Float16* kf = (_Float16*)(ws + (8u << 20));       //  8 MB
  _Float16* vt = (_Float16*)(ws + (16u << 20));      //  8 MB (transposed V)
  _Float16* e  = (_Float16*)(ws + (24u << 20));      //  ~268 KB (+pad)
  _Float16* ao = (_Float16*)(ws + (25u << 20));      //  8 MB
  float* out = (float*)d_out;

  cvt_e_kernel<<<(EROWS * DHc + 255) / 256, 256, 0, stream>>>(rel, e);
  qkv_proj_kernel<<<dim3(32, 8, 3), 256, 0, stream>>>(
      query, key, value, Wq, bq, Wk, bk, Wv, bv, qf, kf, vt);
  flash_attn_kernel<<<dim3(16, 32), 256, 0, stream>>>(qf, kf, vt, e, ao);
  out_proj_kernel<<<dim3(32, 8), 256, 0, stream>>>(ao, Wo, bo, out);
}
